// AFNO3D_34170759807235
// MI455X (gfx1250) — matrix operand order repaired and checked by reading, not compiled, not run on hardware
//
#include <hip/hip_runtime.h>
#include <hip/hip_bf16.h>

// ---------------------------------------------------------------------------
// AFNO3D for MI455X (gfx1250, wave32, WMMA)
//   x: (4,32,32,32,256) f32.  Spectral buffer X: (4,32,32,4,256) complex f32
//   laid out as float2 with index ((b*32+h)*32+w)*4+m)*256 + c.
// Pipeline: pack weights -> T-DFT(32->4) -> FFT_w -> FFT_h -> WMMA MLP
//           -> iFFT_h -> iFFT_w -> iT-DFT(4->32) + residual.
// ---------------------------------------------------------------------------

typedef __attribute__((ext_vector_type(16))) __bf16 v16bf;
typedef __attribute__((ext_vector_type(8)))  float  v8f;

#define AFNO_LAMBDA 0.01f
#define INV_SQRT32  0.17677669529663687f

// cos/sin(2*pi*k/32) twiddle tables: all indices are compile-time after
// unrolling, so these fold to VALU literals (no v_sin/v_cos, no LUT memory).
__device__ static constexpr float CT32[32] = {
     1.00000000f,  0.98078528f,  0.92387953f,  0.83146961f,
     0.70710678f,  0.55557023f,  0.38268343f,  0.19509032f,
     0.00000000f, -0.19509032f, -0.38268343f, -0.55557023f,
    -0.70710678f, -0.83146961f, -0.92387953f, -0.98078528f,
    -1.00000000f, -0.98078528f, -0.92387953f, -0.83146961f,
    -0.70710678f, -0.55557023f, -0.38268343f, -0.19509032f,
     0.00000000f,  0.19509032f,  0.38268343f,  0.55557023f,
     0.70710678f,  0.83146961f,  0.92387953f,  0.98078528f };
__device__ static constexpr float ST32[32] = {
     0.00000000f,  0.19509032f,  0.38268343f,  0.55557023f,
     0.70710678f,  0.83146961f,  0.92387953f,  0.98078528f,
     1.00000000f,  0.98078528f,  0.92387953f,  0.83146961f,
     0.70710678f,  0.55557023f,  0.38268343f,  0.19509032f,
     0.00000000f, -0.19509032f, -0.38268343f, -0.55557023f,
    -0.70710678f, -0.83146961f, -0.92387953f, -0.98078528f,
    -1.00000000f, -0.98078528f, -0.92387953f, -0.83146961f,
    -0.70710678f, -0.55557023f, -0.38268343f, -0.19509032f };

// Native f32 -> bf16 convert (RTNE on gfx1250); avoids bfe/add3 bit games.
static __device__ __forceinline__ __bf16 f2bf(float f) { return (__bf16)f; }

// --------------------------- weight pre-pack -------------------------------
// Wc1 (64x128) = [[W1r, W1i],[-W1i, W1r]]   -> bf16 B-fragment order
// Wc2 (128x64) = [[W2r, W2i],[-W2i, W2r]]   -> bf16 B-fragment order
// B-fragment layout (32x16 / K=32 slab): lane = {n_col (0..15), khalf},
// element e -> K = khalf*16 + e.
__global__ void __launch_bounds__(256)
afno_pack_w(const float* __restrict__ w1, const float* __restrict__ w2,
            __bf16* __restrict__ pw) {
    int gid = blockIdx.x * 256 + threadIdx.x;       // 131072 total
    int n = gid >> 14;
    int r = gid & 16383;
    float val;
    if (r < 8192) {                                  // Wc1 region
        int e = r & 15, lane = (r >> 4) & 31, f = (r >> 9) & 1, t = r >> 10;
        int K = f * 32 + (lane >> 4) * 16 + e;       // 0..63
        int j = t * 16 + (lane & 15);                // 0..127
        int  k   = (K < 32) ? K : K - 32;
        bool hiK = (K >= 32), hiJ = (j >= 64);
        int  jj  = hiJ ? j - 64 : j;
        int  pl  = (hiK != hiJ) ? 1 : 0;             // 0 -> w1r plane, 1 -> w1i
        float sg = (hiK && !hiJ) ? -1.0f : 1.0f;
        val = sg * w1[(((size_t)pl * 8 + n) * 32 + k) * 64 + jj];
    } else {                                         // Wc2 region
        int r2 = r - 8192;
        int e = r2 & 15, lane = (r2 >> 4) & 31, f = (r2 >> 9) & 3, t = r2 >> 11;
        int K = f * 32 + (lane >> 4) * 16 + e;       // 0..127
        int j = t * 16 + (lane & 15);                // 0..63
        int  k   = (K < 64) ? K : K - 64;
        bool hiK = (K >= 64), hiJ = (j >= 32);
        int  jj  = hiJ ? j - 32 : j;
        int  pl  = (hiK != hiJ) ? 1 : 0;
        float sg = (hiK && !hiJ) ? -1.0f : 1.0f;
        val = sg * w2[(((size_t)pl * 8 + n) * 64 + k) * 32 + jj];
    }
    pw[gid] = f2bf(val);
}

// ------------------------ forward T-axis partial DFT -----------------------
// X[b,h,w,m,c] = (1/sqrt32) * sum_t x[b,h,w,t,c] * exp(-2*pi*i*m*t/32), m=0..3
__global__ void __launch_bounds__(256)
afno_tdft(const float* __restrict__ x, float2* __restrict__ X) {
    int id = blockIdx.x * 256 + threadIdx.x;        // 1,048,576 = 4096 spat * 256 c
    int c = id & 255, spat = id >> 8;
    const float* xp = x + (size_t)spat * 8192 + c;
    float a0r = 0.f, a1r = 0.f, a1i = 0.f, a2r = 0.f, a2i = 0.f, a3r = 0.f, a3i = 0.f;
#pragma unroll
    for (int t = 0; t < 32; ++t) {
        float v = xp[(size_t)t * 256];
        const int i1 = t & 31, i2 = (2 * t) & 31, i3 = (3 * t) & 31;
        a0r += v;
        a1r += v * CT32[i1]; a1i -= v * ST32[i1];
        a2r += v * CT32[i2]; a2i -= v * ST32[i2];
        a3r += v * CT32[i3]; a3i -= v * ST32[i3];
    }
    float2* out = X + (size_t)spat * 1024 + c;
    const float s = INV_SQRT32;
    out[0]   = make_float2(a0r * s, 0.f);
    out[256] = make_float2(a1r * s, a1i * s);
    out[512] = make_float2(a2r * s, a2i * s);
    out[768] = make_float2(a3r * s, a3i * s);
}

// ----------------------------- 32-pt line FFT ------------------------------
__host__ __device__ constexpr int rev5(int i) {
    return ((i & 1) << 4) | ((i & 2) << 2) | (i & 4) | ((i & 8) >> 2) | ((i & 16) >> 4);
}

template <int SGN>   // -1 forward, +1 inverse
static __device__ __forceinline__ void fft32(float2* v) {
#pragma unroll
    for (int i = 0; i < 32; ++i) {
        constexpr int R[32] = { rev5(0),rev5(1),rev5(2),rev5(3),rev5(4),rev5(5),rev5(6),rev5(7),
                                rev5(8),rev5(9),rev5(10),rev5(11),rev5(12),rev5(13),rev5(14),rev5(15),
                                rev5(16),rev5(17),rev5(18),rev5(19),rev5(20),rev5(21),rev5(22),rev5(23),
                                rev5(24),rev5(25),rev5(26),rev5(27),rev5(28),rev5(29),rev5(30),rev5(31) };
        int j = R[i];
        if (j > i) { float2 t = v[i]; v[i] = v[j]; v[j] = t; }
    }
#pragma unroll
    for (int s = 1; s <= 5; ++s) {
        const int len = 1 << s, half = len >> 1, step = 32 >> s;
#pragma unroll
        for (int i = 0; i < 32; i += len) {
#pragma unroll
            for (int k = 0; k < half; ++k) {
                const int idx = (k * step) & 31;
                const float cs = CT32[idx];
                const float sn = (SGN < 0) ? -ST32[idx] : ST32[idx];
                float2 u = v[i + k], w = v[i + k + half];
                float2 t = make_float2(cs * w.x - sn * w.y, cs * w.y + sn * w.x);
                v[i + k]        = make_float2(u.x + t.x, u.y + t.y);
                v[i + k + half] = make_float2(u.x - t.x, u.y - t.y);
            }
        }
    }
}

// axis 0 = W, axis 1 = H. ortho 1/sqrt(32) each direction.
template <int SGN>
__global__ void __launch_bounds__(256)
afno_fft_lines(float2* __restrict__ X, int axis) {
    int id = blockIdx.x * 256 + threadIdx.x;        // 131072 lines
    int c = id & 255, m = (id >> 8) & 3, o2 = id >> 10;   // o2: 0..127
    size_t base, stride;
    if (axis == 0) {                                 // FFT along W; o2 = b*32+h
        base = (size_t)o2 * 32768 + (size_t)m * 256 + c;  stride = 1024;
    } else {                                         // FFT along H; o2 = b*32+w
        base = (size_t)(o2 >> 5) * 1048576 + (size_t)(o2 & 31) * 1024
             + (size_t)m * 256 + c;                        stride = 32768;
    }
    float2 v[32];
#pragma unroll
    for (int i = 0; i < 32; ++i) v[i] = X[base + (size_t)i * stride];
    fft32<SGN>(v);
    const float s = INV_SQRT32;
#pragma unroll
    for (int i = 0; i < 32; ++i)
        X[base + (size_t)i * stride] = make_float2(v[i].x * s, v[i].y * s);
}

// --------------------------- spectral MLP (WMMA) ---------------------------
// Per wave: 16 positions x one channel block n.
//  layer1: U(16x64 bf16) x Wc1(64x128)  -> 8 N-tiles, 2 WMMA each
//  layer2: V(16x128 bf16) x Wc2(128x64) -> 4 N-tiles, 4 WMMA each
__global__ void __launch_bounds__(256)
afno_mlp_wmma(float2* __restrict__ X, const __bf16* __restrict__ pw,
              const float* __restrict__ b1, const float* __restrict__ b2) {
    __shared__ __bf16 strans[8][16 * 128];          // per-wave 16x128 bf16 transpose tile
    const int lane  = threadIdx.x & 31;
    const int wave  = threadIdx.x >> 5;
    const int n     = blockIdx.y;                    // channel block 0..7
    const int tileBase = (blockIdx.x * 8 + wave) * 16;
    const int mrow  = lane & 15;                     // A-row / B,D-column within tile
    const int klo   = (lane >> 4) * 8;               // D-layout row base; A-layout K base is 2*klo

    // ---- A1 fragments from spectral buffer (f32 -> bf16): K'=0..31 real, 32..63 imag
    v16bf aLo, aHi;
    const float2* xrow = X + (size_t)(tileBase + mrow) * 256 + n * 32;
#pragma unroll
    for (int e = 0; e < 16; ++e) {
        int k = 2 * klo + e;                         // fragment-internal K, 0..31
        float2 v = xrow[k];
        aLo[e] = f2bf(v.x);
        aHi[e] = f2bf(v.y);
    }

    const __bf16* pw1 = pw + (size_t)n * 16384;
    const __bf16* pw2 = pw1 + 8192;

    // ---- layer 1
    v8f acc[8];
#pragma unroll
    for (int t = 0; t < 8; ++t) {
        v16bf bLo = *(const v16bf*)(pw1 + ((size_t)(t * 2 + 0) * 32 + lane) * 16);
        v16bf bHi = *(const v16bf*)(pw1 + ((size_t)(t * 2 + 1) * 32 + lane) * 16);
        v8f c = {};
        c = __builtin_amdgcn_wmma_f32_16x16x32_bf16(false, aLo, false, bLo, (short)0, c, false, false);
        c = __builtin_amdgcn_wmma_f32_16x16x32_bf16(false, aHi, false, bHi, (short)0, c, false, false);
        acc[t] = c;
    }

    // ---- bias + relu, transpose D-layout -> A-layout via per-wave LDS tile
    __bf16* sm = &strans[wave][0];
#pragma unroll
    for (int t = 0; t < 8; ++t) {
        int j = t * 16 + mrow;                       // output column 0..127
        float bias = (j < 64) ? b1[(size_t)n * 128 + j]
                              : b1[(size_t)(8 + n) * 128 + (j - 64)];
#pragma unroll
        for (int r = 0; r < 8; ++r) {
            int M = klo + r;                         // row 0..15
            float v = fmaxf(acc[t][r] + bias, 0.0f);
            sm[M * 128 + j] = f2bf(v);
        }
    }

    // ---- A2 fragments from LDS (same-wave LDS ops are in order; no barrier)
    v16bf a2[4];
#pragma unroll
    for (int f = 0; f < 4; ++f) {
#pragma unroll
        for (int e = 0; e < 16; ++e) {
            int kint = 2 * klo + e;
            a2[f][e] = sm[mrow * 128 + f * 32 + kint];
        }
    }

    // ---- layer 2 + bias + softshrink + in-place scatter store
#pragma unroll
    for (int t = 0; t < 4; ++t) {
        v8f c = {};
#pragma unroll
        for (int f = 0; f < 4; ++f) {
            v16bf b = *(const v16bf*)(pw2 + ((size_t)(t * 4 + f) * 32 + lane) * 16);
            c = __builtin_amdgcn_wmma_f32_16x16x32_bf16(false, a2[f], false, b, (short)0, c, false, false);
        }
        int j = t * 16 + mrow;                       // 0..63: real | imag halves
        float bias = (j < 32) ? b2[(size_t)n * 32 + j]
                              : b2[(size_t)(8 + n) * 32 + (j - 32)];
        int  ch  = (j < 32) ? j : j - 32;
        int  ri  = (j < 32) ? 0 : 1;
#pragma unroll
        for (int r = 0; r < 8; ++r) {
            int M = klo + r;
            float v = c[r] + bias;
            v = copysignf(fmaxf(fabsf(v) - AFNO_LAMBDA, 0.0f), v);
            float* dst = (float*)(X + (size_t)(tileBase + M) * 256 + n * 32 + ch);
            dst[ri] = v;
        }
    }
}

// -------------------- inverse T-axis DFT + residual add --------------------
// out[t] = x[t] + (1/sqrt32)*( Re Y0 + 2*sum_{m=1..3} (Re Ym cos - Im Ym sin) )
__global__ void __launch_bounds__(256)
afno_itdft_bias(const float* __restrict__ x, const float2* __restrict__ X,
                float* __restrict__ out) {
    int id = blockIdx.x * 256 + threadIdx.x;
    int c = id & 255, spat = id >> 8;
    const float2* yp = X + (size_t)spat * 1024 + c;
    float2 y0 = yp[0], y1 = yp[256], y2 = yp[512], y3 = yp[768];
    const float* xp = x + (size_t)spat * 8192 + c;
    float* op = out + (size_t)spat * 8192 + c;
    const float s = INV_SQRT32;
#pragma unroll
    for (int t = 0; t < 32; ++t) {
        const int i1 = t & 31, i2 = (2 * t) & 31, i3 = (3 * t) & 31;
        float v = y0.x + 2.f * (y1.x * CT32[i1] - y1.y * ST32[i1] +
                                y2.x * CT32[i2] - y2.y * ST32[i2] +
                                y3.x * CT32[i3] - y3.y * ST32[i3]);
        op[(size_t)t * 256] = xp[(size_t)t * 256] + s * v;
    }
}

// ---------------------------------------------------------------------------
extern "C" void kernel_launch(void* const* d_in, const int* in_sizes, int n_in,
                              void* d_out, int out_size, void* d_ws, size_t ws_size,
                              hipStream_t stream) {
    const float* x  = (const float*)d_in[0];
    const float* w1 = (const float*)d_in[1];
    const float* b1 = (const float*)d_in[2];
    const float* w2 = (const float*)d_in[3];
    const float* b2 = (const float*)d_in[4];
    float* out = (float*)d_out;

    // workspace: spectral buffer (33,554,432 B) + packed bf16 weights (262,144 B)
    float2* X  = (float2*)d_ws;
    __bf16* pw = (__bf16*)((char*)d_ws + 33554432u);

    afno_pack_w       <<<512,  256, 0, stream>>>(w1, w2, pw);
    afno_tdft         <<<4096, 256, 0, stream>>>(x, X);
    afno_fft_lines<-1><<<512,  256, 0, stream>>>(X, 0);   // FFT along W
    afno_fft_lines<-1><<<512,  256, 0, stream>>>(X, 1);   // FFT along H
    afno_mlp_wmma     <<<dim3(128, 8), 256, 0, stream>>>(X, pw, b1, b2);
    afno_fft_lines<+1><<<512,  256, 0, stream>>>(X, 1);   // iFFT along H
    afno_fft_lines<+1><<<512,  256, 0, stream>>>(X, 0);   // iFFT along W
    afno_itdft_bias   <<<4096, 256, 0, stream>>>(x, X, out);
}
